// CausalSelfAttention_36593121361921
// MI455X (gfx1250) — matrix operand order repaired and checked by reading, not compiled, not run on hardware
//
#include <hip/hip_runtime.h>
#include <cstdint>

// Sizes from the reference
#define B_    8
#define T_    1024
#define C_    768
#define H_    12
#define D_    64
#define NQKV  2304   // 3*C
#define ROWS  8192   // B*T

typedef __attribute__((ext_vector_type(16))) __bf16 v16bf;
typedef __attribute__((ext_vector_type(8)))  float  v8f;

union FragU { uint4 u[2]; v16bf v; };

__device__ __forceinline__ unsigned short f2bf(float f) {
  union { float f; unsigned int u; } c; c.f = f;
  unsigned int u = c.u;
  u += 0x7FFFu + ((u >> 16) & 1u);     // round-to-nearest-even
  return (unsigned short)(u >> 16);
}

__device__ __forceinline__ uint4 ld16(const unsigned short* p) {
  return *reinterpret_cast<const uint4*>(p);
}

__device__ __forceinline__ v8f vzero8() {
  v8f z;
#pragma unroll
  for (int i = 0; i < 8; ++i) z[i] = 0.0f;
  return z;
}

__device__ __forceinline__ v8f wmma_bf16(v16bf a, v16bf b, v8f c) {
  // D = A(16x32 bf16) * B(32x16 bf16) + C(16x16 f32)
  return __builtin_amdgcn_wmma_f32_16x16x32_bf16(false, a, false, b, (short)0, c,
                                                 false, false);
}

// ---------------------------------------------------------------- converts
__global__ void k_f32_to_bf16(const float* __restrict__ src,
                              unsigned short* __restrict__ dst, int n) {
  for (int i = blockIdx.x * blockDim.x + threadIdx.x; i < n;
       i += gridDim.x * blockDim.x)
    dst[i] = f2bf(src[i]);
}

// src[K][N] (row-major) -> dst[N][K] bf16 (store W^T so B-fragments load contiguously)
__global__ void k_transpose_bf16(const float* __restrict__ src,
                                 unsigned short* __restrict__ dst, int K, int N) {
  int tot = K * N;
  for (int i = blockIdx.x * blockDim.x + threadIdx.x; i < tot;
       i += gridDim.x * blockDim.x) {
    int k = i / N, n = i - k * N;
    dst[(size_t)n * K + k] = f2bf(src[i]);
  }
}

// ---------------------------------------------------------------- QKV GEMM
// x_bf [8192][768] @ W^T_bf [2304][768] + bias -> Q,K [B,H,T,D] bf16 ; V^T [B,H,D,T] bf16
// Wave-tile: 32 (M) x 64 (N): 2 A-frags x 4 B-frags -> 8 WMMAs per 32-wide K step.
__global__ __launch_bounds__(256) void k_qkv_gemm(
    const unsigned short* __restrict__ xbf,
    const unsigned short* __restrict__ wT,
    const float* __restrict__ bias,
    unsigned short* __restrict__ Qb,
    unsigned short* __restrict__ Kb,
    unsigned short* __restrict__ Vt) {
  const int lane = threadIdx.x & 31;
  const int wave = threadIdx.x >> 5;
  const int wt   = blockIdx.x * 8 + wave;       // 9216 wave-tiles total
  const int mtile = wt / 36;                    // 256 M-tiles of 32 rows
  const int ntile = wt % 36;                    // 36 N-tiles of 64 cols
  const int half = lane >> 4, l15 = lane & 15;
  const int kbA = half * 8;                     // A frag K chunk base
  const int kbB = half * 8;                     // B frag K chunk base
  const int n0 = ntile * 64;

  const unsigned short* arow0 = xbf + (size_t)(mtile * 32 + l15) * C_;
  const unsigned short* arow1 = arow0 + (size_t)16 * C_;
  const unsigned short* bbase = wT + (size_t)(n0 + l15) * C_;

  v8f acc[2][4];
#pragma unroll
  for (int i = 0; i < 2; ++i)
#pragma unroll
    for (int j = 0; j < 4; ++j) acc[i][j] = vzero8();

  for (int k0 = 0; k0 < C_; k0 += 32) {
    FragU a0, a1;
    a0.u[0] = ld16(arow0 + k0 + kbA);
    a0.u[1] = ld16(arow0 + k0 + kbA + 16);
    a1.u[0] = ld16(arow1 + k0 + kbA);
    a1.u[1] = ld16(arow1 + k0 + kbA + 16);
    if (k0 + 64 < C_) {   // speculative prefetch two K-steps ahead (global_prefetch_b8)
      __builtin_prefetch(arow0 + k0 + 64, 0, 0);
      __builtin_prefetch(bbase + k0 + 64, 0, 0);
    }
#pragma unroll
    for (int j = 0; j < 4; ++j) {
      const unsigned short* br = bbase + (size_t)j * 16 * C_ + k0 + kbB;
      FragU b;
      b.u[0] = ld16(br);
      b.u[1] = ld16(br + 16);
      acc[0][j] = wmma_bf16(a0.v, b.v, acc[0][j]);
      acc[1][j] = wmma_bf16(a1.v, b.v, acc[1][j]);
    }
  }

#pragma unroll
  for (int j = 0; j < 4; ++j) {
    const int col = n0 + j * 16 + l15;
    const float bv = bias[col];
#pragma unroll
    for (int i = 0; i < 2; ++i) {
#pragma unroll
      for (int r = 0; r < 8; ++r) {
        const float v = acc[i][j][r] + bv;
        const int row = mtile * 32 + i * 16 + r + half * 8;  // token row [0,8192)
        const int b = row >> 10, t = row & 1023;
        const unsigned short bf = f2bf(v);
        if (col < C_) {
          const int h = col >> 6, d = col & 63;
          Qb[(((size_t)b * H_ + h) * T_ + t) * D_ + d] = bf;
        } else if (col < 2 * C_) {
          const int f = col - C_, h = f >> 6, d = f & 63;
          Kb[(((size_t)b * H_ + h) * T_ + t) * D_ + d] = bf;
        } else {
          const int f = col - 2 * C_, h = f >> 6, d = f & 63;
          Vt[(((size_t)b * H_ + h) * D_ + d) * T_ + t] = bf;  // transposed for PV
        }
      }
    }
  }
}

// ---------------------------------------------------------------- flash attention
// One wave = 32 query rows (two 16-row tiles sharing K/V fragments);
// block = 4 waves = 128 query rows of one (b,h).
#define PSTR 40   // padded LDS row stride in halfs (80B -> conflict-free b128 reads)
__global__ __launch_bounds__(128) void k_attn(
    const unsigned short* __restrict__ Qb,
    const unsigned short* __restrict__ Kb,
    const unsigned short* __restrict__ Vt,
    unsigned short* __restrict__ Y) {        // [B,T,C] bf16 (head-merged)
  __shared__ unsigned short plds[4][32 * PSTR];

  const int lane = threadIdx.x & 31;
  const int wave = threadIdx.x >> 5;
  const int qblk = blockIdx.x & 7;           // T/128 = 8 blocks per (b,h)
  const int bh   = blockIdx.x >> 3;
  const int b = bh / H_, h = bh % H_;
  const int qbase = qblk * 128 + wave * 32;
  const int half = lane >> 4, l15 = lane & 15;
  const int kbA = half * 16, kbB = half * 16;

  // Q fragments: 2 row-tiles x (D=0..31, D=32..63), loaded once
  FragU qa[2], qc[2];
#pragma unroll
  for (int i = 0; i < 2; ++i) {
    const unsigned short* qp = Qb + ((size_t)bh * T_ + qbase + i * 16 + l15) * D_;
    qa[i].u[0] = ld16(qp + kbA);      qa[i].u[1] = ld16(qp + kbA + 8);
    qc[i].u[0] = ld16(qp + 32 + kbA); qc[i].u[1] = ld16(qp + 32 + kbA + 8);
  }

  v8f acc[2][4];
#pragma unroll
  for (int i = 0; i < 2; ++i)
#pragma unroll
    for (int j = 0; j < 4; ++j) acc[i][j] = vzero8();
  float mrow[2][8], lrow[2][8];
#pragma unroll
  for (int i = 0; i < 2; ++i)
#pragma unroll
    for (int r = 0; r < 8; ++r) { mrow[i][r] = -INFINITY; lrow[i][r] = 0.0f; }

  const float scale = 0.125f;                // 1/sqrt(64)
  const int kvend = qbase + 32;              // keys <= qbase+31 needed (causal)

  for (int kv = 0; kv < kvend; kv += 32) {
    // ---- K fragments for 32 keys (shared by both Q row-tiles)
    const unsigned short* kr0 = Kb + ((size_t)bh * T_ + kv + l15) * D_;
    const unsigned short* kr1 = kr0 + 16 * D_;
    FragU b0a, b0c, b1a, b1c;
    b0a.u[0] = ld16(kr0 + kbB);      b0a.u[1] = ld16(kr0 + kbB + 8);
    b0c.u[0] = ld16(kr0 + 32 + kbB); b0c.u[1] = ld16(kr0 + 32 + kbB + 8);
    b1a.u[0] = ld16(kr1 + kbB);      b1a.u[1] = ld16(kr1 + kbB + 8);
    b1c.u[0] = ld16(kr1 + 32 + kbB); b1c.u[1] = ld16(kr1 + 32 + kbB + 8);

#pragma unroll
    for (int i = 0; i < 2; ++i) {
      v8f s0 = vzero8(), s1 = vzero8();
      s0 = wmma_bf16(qa[i].v, b0a.v, s0); s0 = wmma_bf16(qc[i].v, b0c.v, s0);
      s1 = wmma_bf16(qa[i].v, b1a.v, s1); s1 = wmma_bf16(qc[i].v, b1c.v, s1);

      // ---- online softmax over this 16x32 tile (rows striped per C-layout)
      float corr[8];
#pragma unroll
      for (int r = 0; r < 8; ++r) {
        const int qrow = qbase + i * 16 + r + half * 8;
        float a0 = s0[r] * scale;
        float a1 = s1[r] * scale;
        if (kv + l15 > qrow)      a0 = -INFINITY;
        if (kv + 16 + l15 > qrow) a1 = -INFINITY;
        float t = fmaxf(a0, a1);
        t = fmaxf(t, __shfl_xor(t, 1, 32));
        t = fmaxf(t, __shfl_xor(t, 2, 32));
        t = fmaxf(t, __shfl_xor(t, 4, 32));
        t = fmaxf(t, __shfl_xor(t, 8, 32));   // reduce within 16-lane half
        const float mn = fmaxf(mrow[i][r], t);
        const float c = __expf(mrow[i][r] - mn);
        mrow[i][r] = mn;
        const float p0 = __expf(a0 - mn);
        const float p1 = __expf(a1 - mn);
        float s = p0 + p1;
        s += __shfl_xor(s, 1, 32);
        s += __shfl_xor(s, 2, 32);
        s += __shfl_xor(s, 4, 32);
        s += __shfl_xor(s, 8, 32);
        lrow[i][r] = lrow[i][r] * c + s;
        corr[r] = c;
        plds[wave][(i * 16 + r + half * 8) * PSTR + l15]      = f2bf(p0);
        plds[wave][(i * 16 + r + half * 8) * PSTR + 16 + l15] = f2bf(p1);
      }
#pragma unroll
      for (int j = 0; j < 4; ++j)
#pragma unroll
        for (int r = 0; r < 8; ++r) acc[i][j][r] *= corr[r];
    }

    // ---- reload P tiles as A-fragments (16x32 each) from LDS
    FragU pf[2];
#pragma unroll
    for (int i = 0; i < 2; ++i) {
      const unsigned short* pl = &plds[wave][(i * 16 + l15) * PSTR + kbA];
      pf[i].u[0] = *reinterpret_cast<const uint4*>(pl);
      pf[i].u[1] = *reinterpret_cast<const uint4*>(pl + 8);
    }

    // ---- O += P V  (4 D-tiles of 16; each V fragment feeds both row-tiles)
#pragma unroll
    for (int j = 0; j < 4; ++j) {
      const unsigned short* vr =
          Vt + ((size_t)bh * D_ + j * 16 + l15) * T_ + kv;
      FragU vf;
      vf.u[0] = ld16(vr + kbB);
      vf.u[1] = ld16(vr + kbB + 8);
      acc[0][j] = wmma_bf16(pf[0].v, vf.v, acc[0][j]);
      acc[1][j] = wmma_bf16(pf[1].v, vf.v, acc[1][j]);
    }
  }

  // ---- normalize and store to head-merged [B,T,C] bf16
#pragma unroll
  for (int i = 0; i < 2; ++i) {
#pragma unroll
    for (int r = 0; r < 8; ++r) {
      const float inv = 1.0f / lrow[i][r];
      const int qrow = qbase + i * 16 + r + half * 8;
      const size_t rowoff = ((size_t)b * T_ + qrow) * C_ + h * D_;
#pragma unroll
      for (int j = 0; j < 4; ++j)
        Y[rowoff + j * 16 + l15] = f2bf(acc[i][j][r] * inv);
    }
  }
}

// ---------------------------------------------------------------- projection
// y_bf [8192][768] @ Wp^T_bf [768][768] + bias -> out fp32 [8192][768]
// Wave-tile: 32 (M) x 64 (N).
__global__ __launch_bounds__(256) void k_proj_gemm(
    const unsigned short* __restrict__ ybf,
    const unsigned short* __restrict__ wT,
    const float* __restrict__ bias,
    float* __restrict__ out) {
  const int lane = threadIdx.x & 31;
  const int wave = threadIdx.x >> 5;
  const int wt   = blockIdx.x * 8 + wave;     // 3072 wave-tiles
  const int mtile = wt / 12;                  // 256 M-tiles of 32 rows
  const int ntile = wt % 12;
  const int half = lane >> 4, l15 = lane & 15;
  const int kbA = half * 8, kbB = half * 8;
  const int n0 = ntile * 64;

  const unsigned short* arow0 = ybf + (size_t)(mtile * 32 + l15) * C_;
  const unsigned short* arow1 = arow0 + (size_t)16 * C_;
  const unsigned short* bbase = wT + (size_t)(n0 + l15) * C_;

  v8f acc[2][4];
#pragma unroll
  for (int i = 0; i < 2; ++i)
#pragma unroll
    for (int j = 0; j < 4; ++j) acc[i][j] = vzero8();

  for (int k0 = 0; k0 < C_; k0 += 32) {
    FragU a0, a1;
    a0.u[0] = ld16(arow0 + k0 + kbA);
    a0.u[1] = ld16(arow0 + k0 + kbA + 16);
    a1.u[0] = ld16(arow1 + k0 + kbA);
    a1.u[1] = ld16(arow1 + k0 + kbA + 16);
    if (k0 + 64 < C_) {
      __builtin_prefetch(arow0 + k0 + 64, 0, 0);
      __builtin_prefetch(bbase + k0 + 64, 0, 0);
    }
#pragma unroll
    for (int j = 0; j < 4; ++j) {
      const unsigned short* br = bbase + (size_t)j * 16 * C_ + k0 + kbB;
      FragU b;
      b.u[0] = ld16(br);
      b.u[1] = ld16(br + 16);
      acc[0][j] = wmma_bf16(a0.v, b.v, acc[0][j]);
      acc[1][j] = wmma_bf16(a1.v, b.v, acc[1][j]);
    }
  }

#pragma unroll
  for (int j = 0; j < 4; ++j) {
    const int col = n0 + j * 16 + l15;
    const float bv = bias[col];
#pragma unroll
    for (int i = 0; i < 2; ++i)
#pragma unroll
      for (int r = 0; r < 8; ++r) {
        const int row = mtile * 32 + i * 16 + r + half * 8;
        out[(size_t)row * C_ + col] = acc[i][j][r] + bv;
      }
  }
}

// ---------------------------------------------------------------- launch
extern "C" void kernel_launch(void* const* d_in, const int* in_sizes, int n_in,
                              void* d_out, int out_size, void* d_ws, size_t ws_size,
                              hipStream_t stream) {
  (void)in_sizes; (void)n_in; (void)out_size; (void)ws_size;
  const float* x     = (const float*)d_in[0];
  const float* Wqkv  = (const float*)d_in[1];
  const float* bqkv  = (const float*)d_in[2];
  const float* Wproj = (const float*)d_in[3];
  const float* bproj = (const float*)d_in[4];
  float* out = (float*)d_out;

  char* ws = (char*)d_ws;
  size_t off = 0;
  auto take = [&](size_t bytes) {
    void* p = ws + off;
    off += (bytes + 255) & ~(size_t)255;
    return p;
  };
  unsigned short* xbf   = (unsigned short*)take((size_t)ROWS * C_ * 2);       // 12.6 MB
  unsigned short* wqkvT = (unsigned short*)take((size_t)NQKV * C_ * 2);       //  3.5 MB
  unsigned short* wpT   = (unsigned short*)take((size_t)C_ * C_ * 2);         //  1.2 MB
  unsigned short* Qb    = (unsigned short*)take((size_t)B_ * H_ * T_ * D_ * 2);
  unsigned short* Kb    = (unsigned short*)take((size_t)B_ * H_ * T_ * D_ * 2);
  unsigned short* Vt    = (unsigned short*)take((size_t)B_ * H_ * T_ * D_ * 2);
  unsigned short* Ybf   = (unsigned short*)take((size_t)ROWS * C_ * 2);       // ~67.6 MB total

  k_f32_to_bf16<<<4096, 256, 0, stream>>>(x, xbf, ROWS * C_);
  k_transpose_bf16<<<2048, 256, 0, stream>>>(Wqkv, wqkvT, C_, NQKV);
  k_transpose_bf16<<<1024, 256, 0, stream>>>(Wproj, wpT, C_, C_);

  // 256 mtiles * 36 ntiles = 9216 wave-tiles / 8 waves per block
  k_qkv_gemm<<<1152, 256, 0, stream>>>(xbf, wqkvT, bqkv, Qb, Kb, Vt);

  // B*H * (T/128) blocks, 4 waves each (32 query rows per wave)
  k_attn<<<768, 128, 0, stream>>>(Qb, Kb, Vt, Ybf);

  // 256 mtiles * 12 ntiles = 3072 wave-tiles / 8 per block
  k_proj_gemm<<<384, 256, 0, stream>>>(Ybf, wpT, bproj, out);
}
